// HybridGatedLayer_72705206387137
// MI455X (gfx1250) — matrix operand order repaired and checked by reading, not compiled, not run on hardware
//
#include <hip/hip_runtime.h>

#define Bn   2
#define Tn   2048
#define Cn   2048
#define Hn   8
#define KVHn 2
#define DHn  256
#define Mrows (Bn*Tn)
#define NEGV (-1.0e9f)

typedef __attribute__((ext_vector_type(16))) __bf16 v16bf;
typedef __attribute__((ext_vector_type(8)))  float  v8f;

union Frag { v16bf v; unsigned u[8]; };
union U4   { uint4 q; unsigned short e[8]; };

__device__ inline unsigned short f2bf(float f) {
  unsigned u = __float_as_uint(f);
  unsigned r = 0x7FFFu + ((u >> 16) & 1u);
  return (unsigned short)((u + r) >> 16);
}
__device__ inline float bf2f(unsigned short b) {
  return __uint_as_float(((unsigned)b) << 16);
}
__device__ inline v8f zero8() {
  v8f z;
#pragma unroll
  for (int i = 0; i < 8; ++i) z[i] = 0.0f;
  return z;
}

// CDNA5 async global->LDS copy (ASYNCcnt-tracked). lds = LDS byte address
// (flat shared address truncated to 32 bits, per ISA 10.2 aperture rules).
__device__ inline void async_ld_b128(unsigned lds, const void* gaddr) {
  asm volatile("global_load_async_to_lds_b128 %0, %1, off"
               :: "v"(lds), "v"(gaddr) : "memory");
}
__device__ inline void wait_async0() {
#if __has_builtin(__builtin_amdgcn_s_wait_asynccnt)
  __builtin_amdgcn_s_wait_asynccnt(0);
#else
  asm volatile("s_wait_asynccnt 0x0" ::: "memory");
#endif
}
__device__ inline void wait_async2() {
#if __has_builtin(__builtin_amdgcn_s_wait_asynccnt)
  __builtin_amdgcn_s_wait_asynccnt(2);
#else
  asm volatile("s_wait_asynccnt 0x2" ::: "memory");
#endif
}
__device__ inline unsigned lds_addr32(const void* p) {
  return (unsigned)(uintptr_t)p;
}

// ---------------------------------------------------------------- convert
__global__ void cvt_f32_bf16(const float* __restrict__ in,
                             unsigned short* __restrict__ out, int n) {
  int i = blockIdx.x * blockDim.x + threadIdx.x;
  if (i < n) out[i] = f2bf(in[i]);
}

// ---------------------------------------------------------------- GEMM (bf16 WMMA, async double-buffered)
#define BM 128
#define BN 64
#define BK 32
#define ASTR 40
#define BSTR 40

template <bool OUTF32>
__global__ __launch_bounds__(256) void gemm_bf16(
    const unsigned short* __restrict__ A, const unsigned short* __restrict__ Bm,
    float* __restrict__ Cf, unsigned short* __restrict__ Cb,
    int M, int N, int K) {
  __shared__ __align__(16) unsigned short As[2][BM][ASTR];
  __shared__ __align__(16) unsigned short BsT[2][BN][BSTR];
  const int t    = threadIdx.x;
  const int lane = t & 31;
  const int wave = t >> 5;
  const int bm   = blockIdx.x * BM;
  const int bn   = blockIdx.y * BN;
  const int wm   = (wave & 3) * 32;
  const int wn   = (wave >> 2) * 32;
  const int n    = lane & 15;
  const int klo  = (lane < 16) ? 0 : 8;
  const int kbb  = (lane < 16) ? 0 : 16;

  v8f acc[2][2];
#pragma unroll
  for (int i = 0; i < 2; ++i)
#pragma unroll
    for (int j = 0; j < 2; ++j) acc[i][j] = zero8();

  // stage A tile asynchronously (2 x b128 per thread), B tile with manual transpose
  auto stageA = [&](int buf, int kb) {
#pragma unroll
    for (int i = 0; i < 2; ++i) {
      int idx = i * 256 + t;
      int r = idx >> 2;
      int c = (idx & 3) * 8;
      async_ld_b128(lds_addr32(&As[buf][r][c]),
                    A + (size_t)(bm + r) * K + kb + c);
    }
  };
  auto stageB = [&](int buf, int kb) {
    int kr = t >> 3;
    int c  = (t & 7) * 8;
    U4 d;
    d.q = *(const uint4*)(Bm + (size_t)(kb + kr) * N + bn + c);
#pragma unroll
    for (int j = 0; j < 8; ++j) BsT[buf][c + j][kr] = d.e[j];
  };

  const int nsteps = K / BK;
  stageA(0, 0);
  stageB(0, 0);

  for (int ki = 0; ki < nsteps; ++ki) {
    const int cur = ki & 1;
    if (ki + 1 < nsteps) {              // prefetch next tile into other buffer
      stageA(cur ^ 1, (ki + 1) * BK);
      stageB(cur ^ 1, (ki + 1) * BK);
      wait_async2();                    // wait for CURRENT batch (in-order), keep next in flight
    } else {
      wait_async0();
    }
    __syncthreads();

    Frag a[2], b[2];
#pragma unroll
    for (int ti = 0; ti < 2; ++ti) {
      const unsigned short* ap = &As[cur][wm + ti * 16 + n][0];
#pragma unroll
      for (int j = 0; j < 4; ++j) {
        a[ti].u[j]     = *(const unsigned*)(ap + klo + 2 * j);
        a[ti].u[4 + j] = *(const unsigned*)(ap + 16 + klo + 2 * j);
      }
    }
#pragma unroll
    for (int tj = 0; tj < 2; ++tj) {
      const unsigned short* bp = &BsT[cur][wn + tj * 16 + n][kbb];
#pragma unroll
      for (int j = 0; j < 8; ++j) b[tj].u[j] = *(const unsigned*)(bp - kbb + klo + 2 * (j & 3) + 16 * (j >> 2));
    }
#pragma unroll
    for (int ti = 0; ti < 2; ++ti)
#pragma unroll
      for (int tj = 0; tj < 2; ++tj)
        acc[ti][tj] = __builtin_amdgcn_wmma_f32_16x16x32_bf16(
            false, a[ti].v, false, b[tj].v, (short)0, acc[ti][tj], false, false);
    __syncthreads();                    // done reading 'cur' before it is refilled
  }

  const int rlo = (lane >> 4) * 8;
#pragma unroll
  for (int ti = 0; ti < 2; ++ti)
#pragma unroll
    for (int tj = 0; tj < 2; ++tj)
#pragma unroll
      for (int r = 0; r < 8; ++r) {
        int row = bm + wm + ti * 16 + r + rlo;
        int col = bn + wn + tj * 16 + n;
        float val = acc[ti][tj][r];
        if (OUTF32) Cf[(size_t)row * N + col] = val;
        else        Cb[(size_t)row * N + col] = f2bf(val);
      }
}

// ---------------------------------------------------------------- RoPE q (+qp), permute to (B,H,T,DH)
__global__ __launch_bounds__(128) void rope_q_kernel(
    const unsigned short* __restrict__ qraw, unsigned short* __restrict__ qb,
    float* __restrict__ qp) {
  int idx = blockIdx.x;                 // b*H*T
  int tt = idx % Tn;
  int h  = (idx / Tn) % Hn;
  int b  = idx / (Tn * Hn);
  int d  = threadIdx.x;                 // 0..127
  const unsigned short* src = qraw + (size_t)(b * Tn + tt) * (Hn * DHn) + h * DHn;
  float x1 = bf2f(src[d]);
  float x2 = bf2f(src[d + 128]);
  float inv = __expf(-(float)d * (9.210340371976184f / 128.0f));
  float ang = (float)tt * inv;
  float s, c;
  __sincosf(ang, &s, &c);
  float o1 = x1 * c - x2 * s;
  float o2 = x2 * c + x1 * s;
  unsigned short* dst = qb + (size_t)((b * Hn + h) * Tn + tt) * DHn;
  dst[d]       = f2bf(o1);
  dst[d + 128] = f2bf(o2);
  __shared__ float sh[3];
  if (d < 3) sh[d] = o1;
  __syncthreads();
  if (d == 0) {
    const float inv3 = 0.5773502691896258f;
    float dx = (h & 4) ? -inv3 : inv3;
    float dy = (h & 2) ? -inv3 : inv3;
    float dz = (h & 1) ? -inv3 : inv3;
    qp[(size_t)(b * Hn + h) * Tn + tt] = sh[0] * dx + sh[1] * dy + sh[2] * dz;
  }
}

// ---------------------------------------------------------------- RoPE k (+kp for all H), permute
__global__ __launch_bounds__(128) void rope_k_kernel(
    const unsigned short* __restrict__ kraw, unsigned short* __restrict__ kb,
    float* __restrict__ kp) {
  int idx = blockIdx.x;                 // b*KVH*T
  int tt = idx % Tn;
  int kv = (idx / Tn) % KVHn;
  int b  = idx / (Tn * KVHn);
  int d  = threadIdx.x;
  const unsigned short* src = kraw + (size_t)(b * Tn + tt) * (KVHn * DHn) + kv * DHn;
  float x1 = bf2f(src[d]);
  float x2 = bf2f(src[d + 128]);
  float inv = __expf(-(float)d * (9.210340371976184f / 128.0f));
  float ang = (float)tt * inv;
  float s, c;
  __sincosf(ang, &s, &c);
  float o1 = x1 * c - x2 * s;
  float o2 = x2 * c + x1 * s;
  unsigned short* dst = kb + (size_t)((b * KVHn + kv) * Tn + tt) * DHn;
  dst[d]       = f2bf(o1);
  dst[d + 128] = f2bf(o2);
  __shared__ float sh[3];
  if (d < 3) sh[d] = o1;
  __syncthreads();
  if (d == 0) {
    const float inv3 = 0.5773502691896258f;
#pragma unroll
    for (int rep = 0; rep < Hn / KVHn; ++rep) {
      int h = kv * (Hn / KVHn) + rep;
      float dx = (h & 4) ? -inv3 : inv3;
      float dy = (h & 2) ? -inv3 : inv3;
      float dz = (h & 1) ? -inv3 : inv3;
      kp[(size_t)(b * Hn + h) * Tn + tt] = sh[0] * dx + sh[1] * dy + sh[2] * dz;
    }
  }
}

// ---------------------------------------------------------------- V permute (B,T,KVH,DH)->(B,KVH,T,DH)
__global__ void perm_v_kernel(const unsigned short* __restrict__ vraw,
                              unsigned short* __restrict__ vb) {
  int i = blockIdx.x * 256 + threadIdx.x;   // over B*KVH*T*DH
  int d  = i % DHn;
  int tt = (i / DHn) % Tn;
  int kv = (i / (DHn * Tn)) % KVHn;
  int b  = i / (DHn * Tn * KVHn);
  vb[i] = vraw[(size_t)(b * Tn + tt) * (KVHn * DHn) + kv * DHn + d];
}

// ---------------------------------------------------------------- fused dual-path flash attention
__global__ __launch_bounds__(256) void attn_kernel(
    const unsigned short* __restrict__ qb, const unsigned short* __restrict__ kb,
    const unsigned short* __restrict__ vb, const float* __restrict__ qp,
    const float* __restrict__ kp, const float* __restrict__ head_scales,
    unsigned short* __restrict__ attn_out) {
  __shared__ __align__(16) unsigned short Ks[32][264];
  __shared__ __align__(16) unsigned short VsT[256][36];
  __shared__ __align__(16) unsigned short Pstd[2][16][32];
  __shared__ __align__(16) unsigned short Pgeo[2][16][32];
  __shared__ float csS[2][16], csG[2][16], lS[2][16], lG[2][16];

  const int t      = threadIdx.x;
  const int lane   = t & 31;
  const int wave   = t >> 5;
  const int g      = wave >> 2;     // row-tile group (0: rows r0.., 1: rows r0+16..)
  const int wslice = wave & 3;      // DH slice (64 cols each)
  const int hi     = lane >> 4;
  const int n      = lane & 15;
  const int klo    = (lane < 16) ? 0 : 8;
  const int kbb    = (lane < 16) ? 0 : 16;
  const int r0     = blockIdx.x * 32;
  const int bh     = blockIdx.y;
  const int b      = bh / Hn;
  const int h      = bh % Hn;
  const int kv     = h / (Hn / KVHn);

  const size_t qbase = (size_t)bh * Tn * DHn;
  const size_t kbase = (size_t)(b * KVHn + kv) * Tn * DHn;
  const float hsv = head_scales[h];

  v8f accS[4], accG[4];
#pragma unroll
  for (int i = 0; i < 4; ++i) { accS[i] = zero8(); accG[i] = zero8(); }

  Frag  qf[8];
  float qpr[8], mS[8], lSr[8], mG[8], lGr[8];
  if (wslice == 0) {
    const unsigned short* ap = qb + qbase + (size_t)(r0 + g * 16 + n) * DHn;
#pragma unroll
    for (int ks = 0; ks < 8; ++ks)
#pragma unroll
      for (int j = 0; j < 4; ++j) {
        qf[ks].u[j]     = *(const unsigned*)(ap + ks * 32 + klo + 2 * j);
        qf[ks].u[4 + j] = *(const unsigned*)(ap + ks * 32 + 16 + klo + 2 * j);
      }
#pragma unroll
    for (int r = 0; r < 8; ++r) {
      qpr[r] = qp[(size_t)bh * Tn + r0 + g * 16 + r + 8 * hi];
      mS[r] = -3.0e38f; lSr[r] = 0.0f;
      mG[r] = -3.0e38f; lGr[r] = 0.0f;
    }
  }

  for (int s0 = 0; s0 <= r0; s0 += 32) {
    __syncthreads();
    // stage K tile via async global->LDS; V tile transposed via VALU scatter
#pragma unroll
    for (int i = 0; i < 4; ++i) {
      int cidx = i * 256 + t;
      int sr = cidx >> 5;
      int dc = (cidx & 31) * 8;
      async_ld_b128(lds_addr32(&Ks[sr][dc]),
                    kb + kbase + (size_t)(s0 + sr) * DHn + dc);
      U4 vdat;
      vdat.q = *(const uint4*)(vb + kbase + (size_t)(s0 + sr) * DHn + dc);
#pragma unroll
      for (int j = 0; j < 8; ++j) VsT[dc + j][sr] = vdat.e[j];
    }
    wait_async0();
    __syncthreads();

    if (wslice == 0) {
      // --- scores: Q(16x256) @ K^T(256x32) via 16 WMMAs
      v8f s0a = zero8(), s1a = zero8();
#pragma unroll
      for (int ks = 0; ks < 8; ++ks) {
        Frag bk0, bk1;
#pragma unroll
        for (int j = 0; j < 8; ++j) {
          bk0.u[j] = *(const unsigned*)&Ks[n][ks * 32 + klo + 2 * (j & 3) + 16 * (j >> 2)];
          bk1.u[j] = *(const unsigned*)&Ks[16 + n][ks * 32 + klo + 2 * (j & 3) + 16 * (j >> 2)];
        }
        s0a = __builtin_amdgcn_wmma_f32_16x16x32_bf16(false, qf[ks].v, false, bk0.v, (short)0, s0a, false, false);
        s1a = __builtin_amdgcn_wmma_f32_16x16x32_bf16(false, qf[ks].v, false, bk1.v, (short)0, s1a, false, false);
      }
      float kp0 = kp[(size_t)bh * Tn + s0 + n];
      float kp1 = kp[(size_t)bh * Tn + s0 + 16 + n];
#pragma unroll
      for (int r = 0; r < 8; ++r) {
        int trow = r0 + g * 16 + r + 8 * hi;
        int sc0 = s0 + n, sc1 = s0 + 16 + n;
        // ---- standard path
        float v0 = (sc0 <= trow) ? s0a[r] * 0.0625f : NEGV;
        float v1 = (sc1 <= trow) ? s1a[r] * 0.0625f : NEGV;
        float mx = fmaxf(v0, v1);
#pragma unroll
        for (int off = 1; off < 16; off <<= 1) mx = fmaxf(mx, __shfl_xor(mx, off));
        float mnew = fmaxf(mS[r], mx);
        float ce = __expf(mS[r] - mnew);
        float p0 = __expf(v0 - mnew), p1 = __expf(v1 - mnew);
        float rs = p0 + p1;
#pragma unroll
        for (int off = 1; off < 16; off <<= 1) rs += __shfl_xor(rs, off);
        lSr[r] = lSr[r] * ce + rs;
        mS[r] = mnew;
        int M = r + 8 * hi;
        Pstd[g][M][n]      = f2bf(p0);
        Pstd[g][M][16 + n] = f2bf(p1);
        if (n == 0) csS[g][M] = ce;
        // ---- geo path (rank-1 scores)
        float g0 = (sc0 <= trow) ? qpr[r] * kp0 : NEGV;
        float g1 = (sc1 <= trow) ? qpr[r] * kp1 : NEGV;
        float gmx = fmaxf(g0, g1);
#pragma unroll
        for (int off = 1; off < 16; off <<= 1) gmx = fmaxf(gmx, __shfl_xor(gmx, off));
        float gmnew = fmaxf(mG[r], gmx);
        float gce = __expf(mG[r] - gmnew);
        float q0 = __expf(g0 - gmnew), q1 = __expf(g1 - gmnew);
        float grs = q0 + q1;
#pragma unroll
        for (int off = 1; off < 16; off <<= 1) grs += __shfl_xor(grs, off);
        lGr[r] = lGr[r] * gce + grs;
        mG[r] = gmnew;
        Pgeo[g][M][n]      = f2bf(q0);
        Pgeo[g][M][16 + n] = f2bf(q1);
        if (n == 0) csG[g][M] = gce;
      }
    }
    __syncthreads();

    // all waves: rescale O accumulators, then P@V (std + geo share V frags)
#pragma unroll
    for (int r = 0; r < 8; ++r) {
      float cs = csS[g][r + 8 * hi];
      float cg = csG[g][r + 8 * hi];
#pragma unroll
      for (int nt = 0; nt < 4; ++nt) { accS[nt][r] *= cs; accG[nt][r] *= cg; }
    }
    Frag pa, ga;
#pragma unroll
    for (int j = 0; j < 4; ++j) {
      pa.u[j]     = *(const unsigned*)&Pstd[g][n][klo + 2 * j];
      pa.u[4 + j] = *(const unsigned*)&Pstd[g][n][16 + klo + 2 * j];
      ga.u[j]     = *(const unsigned*)&Pgeo[g][n][klo + 2 * j];
      ga.u[4 + j] = *(const unsigned*)&Pgeo[g][n][16 + klo + 2 * j];
    }
#pragma unroll
    for (int nt = 0; nt < 4; ++nt) {
      Frag bv;
      int d = wslice * 64 + nt * 16 + n;
#pragma unroll
      for (int j = 0; j < 8; ++j) bv.u[j] = *(const unsigned*)&VsT[d][klo + 2 * (j & 3) + 16 * (j >> 2)];
      accS[nt] = __builtin_amdgcn_wmma_f32_16x16x32_bf16(false, pa.v, false, bv.v, (short)0, accS[nt], false, false);
      accG[nt] = __builtin_amdgcn_wmma_f32_16x16x32_bf16(false, ga.v, false, bv.v, (short)0, accG[nt], false, false);
    }
  }

  if (wslice == 0) {
#pragma unroll
    for (int r = 0; r < 8; ++r)
      if (n == 0) { lS[g][r + 8 * hi] = lSr[r]; lG[g][r + 8 * hi] = lGr[r]; }
  }
  __syncthreads();

#pragma unroll
  for (int r = 0; r < 8; ++r) {
    int M = r + 8 * hi;
    float ls = lS[g][M], lg = lG[g][M];
    int trow = r0 + g * 16 + M;
#pragma unroll
    for (int nt = 0; nt < 4; ++nt) {
      float os = accS[nt][r] / ls;
      float og = accG[nt][r] / lg;
      float o = os + hsv * (og - os);
      int d = wslice * 64 + nt * 16 + n;
      attn_out[(size_t)(b * Tn + trow) * Cn + h * DHn + d] = f2bf(o);
    }
  }
}

// ---------------------------------------------------------------- host
extern "C" void kernel_launch(void* const* d_in, const int* in_sizes, int n_in,
                              void* d_out, int out_size, void* d_ws, size_t ws_size,
                              hipStream_t stream) {
  (void)in_sizes; (void)n_in; (void)out_size; (void)ws_size;
  const float* x  = (const float*)d_in[0];
  const float* Wq = (const float*)d_in[1];
  const float* Wk = (const float*)d_in[2];
  const float* Wv = (const float*)d_in[3];
  const float* Wo = (const float*)d_in[4];
  const float* hs = (const float*)d_in[5];
  float* out = (float*)d_out;

  char* ws = (char*)d_ws;
  size_t off = 0;
  auto alloc = [&](size_t bytes) {
    void* p = ws + off;
    off += (bytes + 255) & ~(size_t)255;
    return p;
  };
  unsigned short* xb   = (unsigned short*)alloc((size_t)Mrows * Cn * 2);
  unsigned short* wqb  = (unsigned short*)alloc((size_t)Cn * Cn * 2);
  unsigned short* wkb  = (unsigned short*)alloc((size_t)Cn * KVHn * DHn * 2);
  unsigned short* wvb  = (unsigned short*)alloc((size_t)Cn * KVHn * DHn * 2);
  unsigned short* wob  = (unsigned short*)alloc((size_t)Cn * Cn * 2);
  unsigned short* qraw = (unsigned short*)alloc((size_t)Mrows * Cn * 2);
  unsigned short* kraw = (unsigned short*)alloc((size_t)Mrows * KVHn * DHn * 2);
  unsigned short* vraw = (unsigned short*)alloc((size_t)Mrows * KVHn * DHn * 2);
  unsigned short* qrb  = (unsigned short*)alloc((size_t)Mrows * Cn * 2);
  unsigned short* krb  = (unsigned short*)alloc((size_t)Mrows * KVHn * DHn * 2);
  unsigned short* vrb  = (unsigned short*)alloc((size_t)Mrows * KVHn * DHn * 2);
  float*          qpv  = (float*)alloc((size_t)Bn * Hn * Tn * 4);
  float*          kpv  = (float*)alloc((size_t)Bn * Hn * Tn * 4);
  unsigned short* attn = (unsigned short*)alloc((size_t)Mrows * Cn * 2);

  // 1) convert inputs to bf16
  {
    int nx = Mrows * Cn;
    cvt_f32_bf16<<<(nx + 255) / 256, 256, 0, stream>>>(x, xb, nx);
    int nq = Cn * Cn;
    cvt_f32_bf16<<<(nq + 255) / 256, 256, 0, stream>>>(Wq, wqb, nq);
    int nk = Cn * KVHn * DHn;
    cvt_f32_bf16<<<(nk + 255) / 256, 256, 0, stream>>>(Wk, wkb, nk);
    cvt_f32_bf16<<<(nk + 255) / 256, 256, 0, stream>>>(Wv, wvb, nk);
    cvt_f32_bf16<<<(nq + 255) / 256, 256, 0, stream>>>(Wo, wob, nq);
  }

  // 2) projection GEMMs (bf16 out)
  gemm_bf16<false><<<dim3(Mrows / BM, Cn / BN), 256, 0, stream>>>(
      xb, wqb, nullptr, qraw, Mrows, Cn, Cn);
  gemm_bf16<false><<<dim3(Mrows / BM, (KVHn * DHn) / BN), 256, 0, stream>>>(
      xb, wkb, nullptr, kraw, Mrows, KVHn * DHn, Cn);
  gemm_bf16<false><<<dim3(Mrows / BM, (KVHn * DHn) / BN), 256, 0, stream>>>(
      xb, wvb, nullptr, vraw, Mrows, KVHn * DHn, Cn);

  // 3) RoPE + permute + qp/kp; V permute
  rope_q_kernel<<<Bn * Hn * Tn, 128, 0, stream>>>(qraw, qrb, qpv);
  rope_k_kernel<<<Bn * KVHn * Tn, 128, 0, stream>>>(kraw, krb, kpv);
  perm_v_kernel<<<(Bn * KVHn * Tn * DHn) / 256, 256, 0, stream>>>(vraw, vrb);

  // 4) fused dual-path causal flash attention
  attn_kernel<<<dim3(Tn / 32, Bn * Hn), 256, 0, stream>>>(
      qrb, krb, vrb, qpv, kpv, hs, attn);

  // 5) output projection (f32 out)
  gemm_bf16<true><<<dim3(Mrows / BM, Cn / BN), 256, 0, stream>>>(
      attn, wob, out, nullptr, Mrows, Cn, Cn);
}
